// Moondream3DecoderLayer_8804682957002
// MI455X (gfx1250) — matrix operand order repaired and checked by reading, not compiled, not run on hardware
//
#include <hip/hip_runtime.h>

// ---------------------------------------------------------------------------
// Moondream3 decoder layer for MI455X (gfx1250), wave32 + WMMA f16->f32,
// async global->LDS staging (ASYNCcnt), transposed-weight GEMM tiles.
// B=1, S=2048, H=2048, I=8192, NH=32, NKV=8, HD=64, ROT=32.
// ---------------------------------------------------------------------------

typedef __attribute__((ext_vector_type(16))) _Float16 v16h;
typedef __attribute__((ext_vector_type(8)))  _Float16 v8h;
typedef __attribute__((ext_vector_type(8)))  float    v8f;

#define S_LEN 2048
#define H_DIM 2048
#define I_DIM 8192
#define NH    32
#define NKV   8
#define HD    64
#define QKV_DIM 3072

union V16U { v16h v; v8h h[2]; };

__device__ __forceinline__ v8f wmma_f16(v16h a, v16h b, v8f c) {
  return __builtin_amdgcn_wmma_f32_16x16x32_f16(
      /*neg_a=*/false, a, /*neg_b=*/false, b,
      /*c_mod=*/(short)0, c, /*reuse_a=*/false, /*reuse_b=*/false);
}

// Async global->LDS copy of 16 bytes per lane (GVS addressing).
__device__ __forceinline__ void async_load_b128(unsigned lds_off,
                                                unsigned goff,
                                                const void* base) {
  asm volatile("global_load_async_to_lds_b128 %0, %1, %2"
               :
               : "v"(lds_off), "v"(goff), "s"(base)
               : "memory");
}
__device__ __forceinline__ void wait_async0() {
  asm volatile("s_wait_asynccnt 0x0" ::: "memory");
}
__device__ __forceinline__ void wait_ds0() {
  asm volatile("s_wait_dscnt 0x0" ::: "memory");
}

// -------------------------------- fp32 -> f16 ------------------------------
__global__ void f32_to_f16_kernel(const float* __restrict__ src,
                                  _Float16* __restrict__ dst, size_t n) {
  size_t i = (size_t)blockIdx.x * blockDim.x + threadIdx.x;
  if (i < n) dst[i] = (_Float16)src[i];
}

// fp32 [K][N] -> f16 transposed [N][K] (weights pass; bandwidth-trivial)
__global__ void f32_to_f16_t_kernel(const float* __restrict__ src,
                                    _Float16* __restrict__ dst,
                                    int K, int N) {
  size_t i = (size_t)blockIdx.x * blockDim.x + threadIdx.x;
  if (i >= (size_t)K * N) return;
  int k = (int)(i / N);
  int n = (int)(i - (size_t)k * N);
  dst[(size_t)n * K + k] = (_Float16)src[i];
}

// -------------------------------- LayerNorm --------------------------------
__global__ __launch_bounds__(256) void ln_kernel(
    const float* __restrict__ x, const float* __restrict__ w,
    const float* __restrict__ b, _Float16* __restrict__ xn_h, int H) {
  __shared__ float rs[256], rq[256];
  const int s = blockIdx.x, t = threadIdx.x;
  const float* row = x + (size_t)s * H;
  float sum = 0.f, sq = 0.f;
  for (int j = t; j < H; j += 256) { float v = row[j]; sum += v; sq += v * v; }
  rs[t] = sum; rq[t] = sq;
  __syncthreads();
  for (int o = 128; o > 0; o >>= 1) {
    if (t < o) { rs[t] += rs[t + o]; rq[t] += rq[t + o]; }
    __syncthreads();
  }
  const float mu  = rs[0] / (float)H;
  const float var = rq[0] / (float)H - mu * mu;
  const float inv = rsqrtf(var + 1e-5f);
  for (int j = t; j < H; j += 256) {
    float v = (row[j] - mu) * inv * w[j] + b[j];
    xn_h[(size_t)s * H + j] = (_Float16)v;
  }
}

// ------------- Tiled WMMA GEMM: C(f32) = A(f16,[M][K]) * BT(f16,[N][K]) ----
// Block tile 128x64, K-step 32. 256 threads = 8 waves; wave w owns rows
// w*16..w*16+15 and all 64 cols (4 accumulators, 4 WMMAs per K-step).
// Tiles staged with global_load_async_to_lds_b128 (ASYNCcnt).
#define GBM 128
#define GBN 64
#define GBK 32
__global__ __launch_bounds__(256) void gemm_f16_kernel(
    const _Float16* __restrict__ A, int lda,
    const _Float16* __restrict__ BT, int ldb,
    float* __restrict__ C, int ldc, int K) {
  __shared__ _Float16 sA[GBM][GBK + 8];
  __shared__ _Float16 sB[GBN][GBK + 8];
  const int t    = threadIdx.x;
  const int wave = t >> 5;
  const int lane = t & 31;
  const int bm = blockIdx.y * GBM;
  const int bn = blockIdx.x * GBN;

  const int mrow = wave * 16 + (lane & 15);
  const int kba  = (lane >> 4) * 8;    // A-fragment K base  {0,8}
  const int nn   = lane & 15;
  const int kbb  = (lane >> 4) * 8;    // B-fragment K base  {0,8}

  // staging coordinates
  const int ar = t >> 1;               // 0..127
  const int ac = (t & 1) * 16;         // 0 or 16
  const int br = t >> 2;               // 0..63
  const int bc = (t & 3) * 8;          // 0,8,16,24

  const unsigned ldsA = (unsigned)(uintptr_t)&sA[0][0];
  const unsigned ldsB = (unsigned)(uintptr_t)&sB[0][0];
  const unsigned la0 = ldsA + (unsigned)(ar * (GBK + 8) + ac) * 2u;
  const unsigned lb0 = ldsB + (unsigned)(br * (GBK + 8) + bc) * 2u;
  const size_t   gaRow = (size_t)(bm + ar) * lda;
  const size_t   gbRow = (size_t)(bn + br) * ldb;

  v8f acc[4] = {};

  for (int k0 = 0; k0 < K; k0 += GBK) {
    { // async stage: A two b128 per thread, B one b128 per thread
      unsigned ga = (unsigned)((gaRow + k0 + ac) * 2);
      async_load_b128(la0,        ga,      A);
      async_load_b128(la0 + 16u,  ga + 16, A);
      unsigned gb = (unsigned)((gbRow + k0 + bc) * 2);
      async_load_b128(lb0,        gb,      BT);
      if (k0 + GBK < K) {  // speculative prefetch of next K-tile
        __builtin_prefetch(A + gaRow + k0 + GBK, 0, 0);
        __builtin_prefetch(BT + gbRow + k0 + GBK, 0, 0);
      }
    }
    wait_async0();
    __syncthreads();

    // Preload A + all four B fragments so ds_load_b128s overlap and the
    // compiler can use partial s_wait_dscnt before each WMMA.
    V16U a, b0, b1, b2, b3;
    a.h[0]  = *(const v8h*)&sA[mrow][kba];
    a.h[1]  = *(const v8h*)&sA[mrow][kba + 16];
    b0.h[0] = *(const v8h*)&sB[nn][kbb];
    b0.h[1] = *(const v8h*)&sB[nn][kbb + 16];
    b1.h[0] = *(const v8h*)&sB[16 + nn][kbb];
    b1.h[1] = *(const v8h*)&sB[16 + nn][kbb + 16];
    b2.h[0] = *(const v8h*)&sB[32 + nn][kbb];
    b2.h[1] = *(const v8h*)&sB[32 + nn][kbb + 16];
    b3.h[0] = *(const v8h*)&sB[48 + nn][kbb];
    b3.h[1] = *(const v8h*)&sB[48 + nn][kbb + 16];
    acc[0] = wmma_f16(a.v, b0.v, acc[0]);
    acc[1] = wmma_f16(a.v, b1.v, acc[1]);
    acc[2] = wmma_f16(a.v, b2.v, acc[2]);
    acc[3] = wmma_f16(a.v, b3.v, acc[3]);
    __syncthreads();
  }

  const int rb = bm + wave * 16 + ((lane >> 4) * 8);
  const int c0 = bn + nn;
  #pragma unroll
  for (int tc = 0; tc < 4; ++tc)
    #pragma unroll
    for (int r = 0; r < 8; ++r)
      C[(size_t)(rb + r) * ldc + c0 + tc * 16] = acc[tc][r];
}

// ------------------ tau: tanh(gelu(qkv) @ tau_w{q,v})  ---------------------
__global__ __launch_bounds__(256) void tau_kernel(
    const float* __restrict__ qkv,       // [S][3072]
    const float* __restrict__ tau_wq,    // [3072][32]
    const float* __restrict__ tau_wv,    // [3072][32]
    float* __restrict__ tokq, float* __restrict__ tokv) {  // [S][32]
  __shared__ float g[256];
  const int s = blockIdx.x, t = threadIdx.x;
  const int j = t & 31;
  float accq = 0.f, accv = 0.f;
  for (int k0 = 0; k0 < QKV_DIM; k0 += 256) {
    float x = qkv[(size_t)s * QKV_DIM + k0 + t];
    g[t] = 0.5f * x * (1.f + erff(x * 0.70710678118f));  // exact gelu
    __syncthreads();
    if (t < 64) {
      const float* W = (t < 32) ? tau_wq : tau_wv;
      float a = 0.f;
      for (int kk = 0; kk < 256; ++kk)
        a += g[kk] * W[(size_t)(k0 + kk) * 32 + j];
      if (t < 32) accq += a; else accv += a;
    }
    __syncthreads();
  }
  if (t < 32)      tokq[(size_t)s * 32 + j] = tanhf(accq);
  else if (t < 64) tokv[(size_t)s * 32 + j] = tanhf(accv);
}

// ------ tau scale + RoPE -> f16 qh/kh [head][S][64], vhT [kvh][64][S] ------
__global__ __launch_bounds__(64) void rope_scale_kernel(
    const float* __restrict__ qkv, const float* __restrict__ tokq,
    const float* __restrict__ tokv, const float* __restrict__ tau_alpha,
    const long long* __restrict__ pos_ids,
    const float* __restrict__ cosb, const float* __restrict__ sinb,
    _Float16* __restrict__ qh, _Float16* __restrict__ kh,
    _Float16* __restrict__ vhT) {
  const int s = blockIdx.x;
  const int h = blockIdx.y;   // 0..31
  const int d = threadIdx.x;  // 0..63
  const float logp = logf((float)(pos_ids[s] + 1));
  const float tau_pos = 0.5f + 1.f / (1.f + expf(-tau_alpha[h] * logp));
  const float* row = qkv + (size_t)s * QKV_DIM;

  const float tq = tokq[s * 32 + h] + tau_pos;
  float val;
  if (d < 32) {
    int tp = d >> 1;
    float r = row[h * 64 + tp]      * tq;
    float i = row[h * 64 + tp + 16] * tq;
    float c = cosb[(size_t)s * 16 + tp], sn = sinb[(size_t)s * 16 + tp];
    val = (d & 1) ? (r * sn + i * c) : (r * c - i * sn);
  } else {
    val = row[h * 64 + d] * tq;
  }
  qh[((size_t)h * S_LEN + s) * HD + d] = (_Float16)val;

  if (h < NKV) {
    const float* krow = row + 2048;
    float kv;
    if (d < 32) {
      int tp = d >> 1;
      float r = krow[h * 64 + tp], i = krow[h * 64 + tp + 16];
      float c = cosb[(size_t)s * 16 + tp], sn = sinb[(size_t)s * 16 + tp];
      kv = (d & 1) ? (r * sn + i * c) : (r * c - i * sn);
    } else {
      kv = krow[h * 64 + d];
    }
    kh[((size_t)h * S_LEN + s) * HD + d] = (_Float16)kv;
    const float tv = tokv[s * 32 + h] + tau_pos;
    // transposed store: [kvh][d][S] so PV B-fragments are contiguous in keys
    vhT[((size_t)h * HD + d) * S_LEN + s] =
        (_Float16)(row[2560 + h * 64 + d] * tv);
  }
}

// -------- flash attention: 1 wave per (16-query tile, head), WMMA ----------
__global__ __launch_bounds__(32) void attn_kernel(
    const _Float16* __restrict__ qh, const _Float16* __restrict__ kh,
    const _Float16* __restrict__ vhT, _Float16* __restrict__ attn_h) {
  __shared__ float    sS[16][33];   // score tile, 16 queries x 32 keys
  __shared__ _Float16 sP[16][32];   // softmaxed probs (f16)
  __shared__ float    sScale[16];
  __shared__ float    sInv[16];

  const int qt   = blockIdx.x;
  const int h    = blockIdx.y;
  const int hk   = h >> 2;          // GQA: NH/NKV = 4
  const int lane = threadIdx.x;
  const int m    = lane & 15;
  const int kba  = (lane >> 4) * 8;
  const int n    = lane & 15;
  const int kbb  = (lane >> 4) * 8;

  const _Float16* Q  = qh  + ((size_t)h * S_LEN + qt * 16) * HD;
  const _Float16* K  = kh  + (size_t)hk * S_LEN * HD;
  const _Float16* Vt = vhT + (size_t)hk * HD * S_LEN;

  // Q A-fragments for the two 32-wide d-chunks (paired b128 loads)
  V16U aq0, aq1;
  aq0.h[0] = *(const v8h*)(Q + m * HD + kba);
  aq0.h[1] = *(const v8h*)(Q + m * HD + kba + 16);
  aq1.h[0] = *(const v8h*)(Q + m * HD + 32 + kba);
  aq1.h[1] = *(const v8h*)(Q + m * HD + 32 + kba + 16);

  v8f o0 = {}, o1 = {}, o2 = {}, o3 = {};
  float mrow = -3.0e38f, lrow = 0.f;
  const int qmax = qt * 16 + 15;

  for (int key0 = 0; key0 <= qmax; key0 += 32) {
    // ---- S = Q K^T / 8 : contraction over d (contiguous K rows) ----
    v8f s0 = {}, s1 = {};
    {
      V16U ba, bb, bc, bd;
      const _Float16* kp0 = K + (size_t)(key0 + n) * HD + kbb;
      const _Float16* kp1 = K + (size_t)(key0 + 16 + n) * HD + kbb;
      ba.h[0] = *(const v8h*)kp0;        ba.h[1] = *(const v8h*)(kp0 + 16);
      bb.h[0] = *(const v8h*)(kp0 + 32); bb.h[1] = *(const v8h*)(kp0 + 48);
      bc.h[0] = *(const v8h*)kp1;        bc.h[1] = *(const v8h*)(kp1 + 16);
      bd.h[0] = *(const v8h*)(kp1 + 32); bd.h[1] = *(const v8h*)(kp1 + 48);
      s0 = wmma_f16(aq0.v, ba.v, s0);
      s0 = wmma_f16(aq1.v, bb.v, s0);
      s1 = wmma_f16(aq0.v, bc.v, s1);
      s1 = wmma_f16(aq1.v, bd.v, s1);
    }
    #pragma unroll
    for (int r = 0; r < 8; ++r) {
      int row = r + ((lane >> 4) * 8);
      sS[row][n]      = s0[r] * 0.125f;
      sS[row][16 + n] = s1[r] * 0.125f;
    }
    wait_ds0();

    // ---- online softmax, lane m owns query row m ----
    if (lane < 16) {
      const int qrow = qt * 16 + lane;
      float vals[32];
      float mx = mrow;
      #pragma unroll
      for (int c = 0; c < 32; ++c) {
        float sc = sS[lane][c];
        if (key0 + c > qrow) sc = -1.0e9f;  // causal mask (matches reference)
        vals[c] = sc;
        mx = fmaxf(mx, sc);
      }
      const float scale = expf(mrow - mx);
      float sum = 0.f;
      #pragma unroll
      for (int c = 0; c < 32; ++c) {
        float p = expf(vals[c] - mx);
        sum += p;
        sP[lane][c] = (_Float16)p;
      }
      lrow = lrow * scale + sum;
      mrow = mx;
      sScale[lane] = scale;
    }
    wait_ds0();

    #pragma unroll
    for (int r = 0; r < 8; ++r) {
      float sc = sScale[r + ((lane >> 4) * 8)];
      o0[r] *= sc; o1[r] *= sc; o2[r] *= sc; o3[r] *= sc;
    }

    // ---- O += P V : A = P(16x32) from LDS, B from transposed V ----
    V16U ap, v0, v1, v2, v3;
    ap.h[0] = *(const v8h*)&sP[m][kba];
    ap.h[1] = *(const v8h*)&sP[m][kba + 16];
    {
      const _Float16* vp0 = Vt + (size_t)n * S_LEN + key0 + kbb;
      const _Float16* vp1 = vp0 + (size_t)16 * S_LEN;
      const _Float16* vp2 = vp1 + (size_t)16 * S_LEN;
      const _Float16* vp3 = vp2 + (size_t)16 * S_LEN;
      v0.h[0] = *(const v8h*)vp0; v0.h[1] = *(const v8h*)(vp0 + 16);
      v1.h[0] = *(const v8h*)vp1; v1.h[1] = *(const v8h*)(vp1 + 16);
      v2.h[0] = *(const v8h*)vp2; v2.h[1] = *(const v8h*)(vp2 + 16);
      v3.h[0] = *(const v8h*)vp3; v3.h[1] = *(const v8h*)(vp3 + 16);
      o0 = wmma_f16(ap.v, v0.v, o0);
      o1 = wmma_f16(ap.v, v1.v, o1);
      o2 = wmma_f16(ap.v, v2.v, o2);
      o3 = wmma_f16(ap.v, v3.v, o3);
    }
  }

  if (lane < 16) sInv[lane] = 1.f / lrow;
  wait_ds0();

  #pragma unroll
  for (int r = 0; r < 8; ++r) {
    const int row = r + ((lane >> 4) * 8);
    const float inv = sInv[row];
    const size_t base = (size_t)(qt * 16 + row) * (NH * HD) + h * HD;
    attn_h[base + n]      = (_Float16)(o0[r] * inv);
    attn_h[base + 16 + n] = (_Float16)(o1[r] * inv);
    attn_h[base + 32 + n] = (_Float16)(o2[r] * inv);
    attn_h[base + 48 + n] = (_Float16)(o3[r] * inv);
  }
}

// --------------------------- MLP gelu(tanh approx) -------------------------
__global__ void gelu_up_kernel(const float* __restrict__ up,
                               const float* __restrict__ up_b,
                               _Float16* __restrict__ h16, size_t n) {
  size_t i = (size_t)blockIdx.x * blockDim.x + threadIdx.x;
  if (i >= n) return;
  float x = up[i] + up_b[i % I_DIM];
  float t = tanhf(0.7978845608f * (x + 0.044715f * x * x * x));
  h16[i] = (_Float16)(0.5f * x * (1.f + t));
}

// ------------------------------ final residual -----------------------------
__global__ void final_add_kernel(const float* __restrict__ hidden,
                                 const float* __restrict__ attnout,
                                 const float* __restrict__ mlp,
                                 const float* __restrict__ down_b,
                                 float* __restrict__ out, size_t n) {
  size_t i = (size_t)blockIdx.x * blockDim.x + threadIdx.x;
  if (i >= n) return;
  out[i] = hidden[i] + attnout[i] + mlp[i] + down_b[i % H_DIM];
}

// ---------------------------------------------------------------------------
static inline size_t align256(size_t x) { return (x + 255) & ~(size_t)255; }

extern "C" void kernel_launch(void* const* d_in, const int* in_sizes, int n_in,
                              void* d_out, int out_size, void* d_ws,
                              size_t ws_size, hipStream_t stream) {
  (void)in_sizes; (void)n_in; (void)out_size; (void)ws_size;
  const float*     hidden   = (const float*)d_in[0];
  const long long* pos_ids  = (const long long*)d_in[1];
  const float*     cosb     = (const float*)d_in[2];
  const float*     sinb     = (const float*)d_in[3];
  const float*     ln_w     = (const float*)d_in[4];
  const float*     ln_b     = (const float*)d_in[5];
  const float*     wq       = (const float*)d_in[6];
  const float*     wk       = (const float*)d_in[7];
  const float*     wv       = (const float*)d_in[8];
  const float*     wo       = (const float*)d_in[9];
  const float*     tau_wq   = (const float*)d_in[10];
  const float*     tau_wv   = (const float*)d_in[11];
  const float*     tau_al   = (const float*)d_in[12];
  const float*     up_w     = (const float*)d_in[13];
  const float*     up_b     = (const float*)d_in[14];
  const float*     down_w   = (const float*)d_in[15];
  const float*     down_b   = (const float*)d_in[16];
  float* out = (float*)d_out;

  // ---- workspace carve-up ----
  char* p = (char*)d_ws;
  size_t off = 0;
  auto alloc = [&](size_t bytes) { void* r = p + off; off = align256(off + bytes); return r; };
  _Float16* xn_h   = (_Float16*)alloc((size_t)S_LEN * H_DIM * 2);
  _Float16* wq_t   = (_Float16*)alloc((size_t)H_DIM * 2048 * 2);   // [2048][H]
  _Float16* wk_t   = (_Float16*)alloc((size_t)H_DIM * 512 * 2);    // [512][H]
  _Float16* wv_t   = (_Float16*)alloc((size_t)H_DIM * 512 * 2);    // [512][H]
  _Float16* wo_t   = (_Float16*)alloc((size_t)2048 * H_DIM * 2);   // [H][2048]
  _Float16* up_t   = (_Float16*)alloc((size_t)H_DIM * I_DIM * 2);  // [I][H]
  _Float16* down_t = (_Float16*)alloc((size_t)I_DIM * H_DIM * 2);  // [H][I]
  float*    qkvf   = (float*)alloc((size_t)S_LEN * QKV_DIM * 4);
  float*    tokq   = (float*)alloc((size_t)S_LEN * 32 * 4);
  float*    tokv   = (float*)alloc((size_t)S_LEN * 32 * 4);
  _Float16* qh     = (_Float16*)alloc((size_t)NH  * S_LEN * HD * 2);
  _Float16* khb    = (_Float16*)alloc((size_t)NKV * S_LEN * HD * 2);
  _Float16* vhT    = (_Float16*)alloc((size_t)NKV * HD * S_LEN * 2);
  _Float16* attn_h = (_Float16*)alloc((size_t)S_LEN * 2048 * 2);
  float*    attnout= (float*)alloc((size_t)S_LEN * H_DIM * 4);
  float*    upf    = (float*)alloc((size_t)S_LEN * I_DIM * 4);
  _Float16* h16    = (_Float16*)alloc((size_t)S_LEN * I_DIM * 2);
  float*    mlpf   = (float*)alloc((size_t)S_LEN * H_DIM * 4);

  auto convT = [&](const float* s, _Float16* d, int K, int N) {
    size_t n = (size_t)K * N;
    f32_to_f16_t_kernel<<<dim3((unsigned)((n + 255) / 256)), 256, 0, stream>>>(
        s, d, K, N);
  };
  convT(wq,     wq_t,   H_DIM, 2048);
  convT(wk,     wk_t,   H_DIM, 512);
  convT(wv,     wv_t,   H_DIM, 512);
  convT(wo,     wo_t,   2048,  H_DIM);
  convT(up_w,   up_t,   H_DIM, I_DIM);
  convT(down_w, down_t, I_DIM, H_DIM);

  // LayerNorm -> f16 xn
  ln_kernel<<<dim3(S_LEN), 256, 0, stream>>>(hidden, ln_w, ln_b, xn_h, H_DIM);

  // QKV projections into one [S][3072] fp32 buffer
  gemm_f16_kernel<<<dim3(2048 / GBN, S_LEN / GBM), 256, 0, stream>>>(
      xn_h, H_DIM, wq_t, H_DIM, qkvf, QKV_DIM, H_DIM);
  gemm_f16_kernel<<<dim3(512 / GBN, S_LEN / GBM), 256, 0, stream>>>(
      xn_h, H_DIM, wk_t, H_DIM, qkvf + 2048, QKV_DIM, H_DIM);
  gemm_f16_kernel<<<dim3(512 / GBN, S_LEN / GBM), 256, 0, stream>>>(
      xn_h, H_DIM, wv_t, H_DIM, qkvf + 2560, QKV_DIM, H_DIM);

  // token tau gates
  tau_kernel<<<dim3(S_LEN), 256, 0, stream>>>(qkvf, tau_wq, tau_wv, tokq, tokv);

  // tau scaling + RoPE -> f16 q/k heads, transposed v
  rope_scale_kernel<<<dim3(S_LEN, NH), 64, 0, stream>>>(
      qkvf, tokq, tokv, tau_al, pos_ids, cosb, sinb, qh, khb, vhT);

  // flash attention (causal, GQA)
  attn_kernel<<<dim3(S_LEN / 16, NH), 32, 0, stream>>>(qh, khb, vhT, attn_h);

  // output projection
  gemm_f16_kernel<<<dim3(H_DIM / GBN, S_LEN / GBM), 256, 0, stream>>>(
      attn_h, 2048, wo_t, 2048, attnout, H_DIM, 2048);

  // MLP: up -> gelu(tanh) -> down
  gemm_f16_kernel<<<dim3(I_DIM / GBN, S_LEN / GBM), 256, 0, stream>>>(
      xn_h, H_DIM, up_t, H_DIM, upf, I_DIM, H_DIM);
  {
    size_t n = (size_t)S_LEN * I_DIM;
    gelu_up_kernel<<<dim3((unsigned)((n + 255) / 256)), 256, 0, stream>>>(
        upf, up_b, h16, n);
  }
  gemm_f16_kernel<<<dim3(H_DIM / GBN, S_LEN / GBM), 256, 0, stream>>>(
      h16, I_DIM, down_t, I_DIM, mlpf, H_DIM, I_DIM);

  // residual + biases
  {
    size_t n = (size_t)S_LEN * H_DIM;
    final_add_kernel<<<dim3((unsigned)((n + 255) / 256)), 256, 0, stream>>>(
        hidden, attnout, mlpf, down_b, out, n);
  }
}
